// Scorer2_46471546143413
// MI455X (gfx1250) — compile-verified, matrix operand order checked by reading, not run on hardware
//
#include <hip/hip_runtime.h>
#include <cstdint>
#include <cstddef>

typedef unsigned short u16;
typedef __attribute__((ext_vector_type(16))) __bf16        bf16x16;
typedef __attribute__((ext_vector_type(8)))  float         f32x8;
typedef __attribute__((ext_vector_type(4)))  unsigned int  u32x4;

#define GLOBAL_AS __attribute__((address_space(1)))
typedef const GLOBAL_AS u16*   gcu16;
typedef const GLOBAL_AS u32x4* gcu4;

// ---------- bf16 helpers ----------
__device__ __forceinline__ u16 f32_to_bf16(float f) {
    union { float f; uint32_t u; } c; c.f = f;
    uint32_t r = c.u + 0x7FFFu + ((c.u >> 16) & 1u);   // round-to-nearest-even
    return (u16)(r >> 16);
}
__device__ __forceinline__ float bf16_to_f32(u16 h) {
    union { uint32_t u; float f; } c; c.u = ((uint32_t)h) << 16;
    return c.f;
}

// ---------- WMMA fragment load from a per-lane GLOBAL base pointer ----------
// 16-bit A/B fragment layout (CDNA5 ISA 7.12.2): lanes 0-15 hold row r0+lane,
// K chunks {k..k+7, k+16..k+23}; lanes 16-31 hold the same rows with K chunks
// {k+8..k+15, k+24..k+31}.  `p` already points at (row, k_chunk_base), so the
// two halves are one global_load_b128 at offset 0 and one at byte offset 32.
__device__ __forceinline__ bf16x16 frag_at(gcu16 p) {
    u32x4 lo = *(gcu4)p;
    u32x4 hi = *(gcu4)(p + 16);
    struct { u32x4 a, b; } t = { lo, hi };
    bf16x16 f;
    __builtin_memcpy(&f, &t, sizeof(f));
    return f;
}

__device__ __forceinline__ f32x8 wmma_bf16(bf16x16 a, bf16x16 b, f32x8 c) {
    return __builtin_amdgcn_wmma_f32_16x16x32_bf16(false, a, false, b,
                                                   (short)0, c, false, false);
}

// ---------- convert / pad / (optional transpose) fp32 -> bf16 ----------
__global__ void cvt_pad(const float* __restrict__ src, u16* __restrict__ dst,
                        int srcR, int srcC, int dstR, int dstC, int transpose) {
    int id = blockIdx.x * blockDim.x + threadIdx.x;
    int total = dstR * dstC;
    if (id >= total) return;
    int i = id / dstC, j = id % dstC;
    float v = 0.f;
    if (!transpose) {
        if (i < srcR && j < srcC) v = src[(size_t)i * srcC + j];
    } else {
        if (j < srcR && i < srcC) v = src[(size_t)j * srcC + i];
    }
    dst[id] = f32_to_bf16(v);
}

// ---------- generic bf16 GEMM:  C = act(A[M,K] @ B[Np,K]^T + bias) ----------
// One wave -> one 64x64 output tile via 4x4 grid of 16x16x32 WMMAs.
// Fragment pointers are global-AS, hoisted, and advanced 32 elems per K-step.
__global__ __launch_bounds__(256)
void gemm_bf16(const u16* __restrict__ A, const u16* __restrict__ B,
               const float* __restrict__ bias, int nbias, int do_relu,
               u16* __restrict__ C,
               int M, int Np, int K, int lda, int ldb, int ldc) {
    int wave = (blockIdx.x * blockDim.x + threadIdx.x) >> 5;
    int lane = threadIdx.x & 31;
    int tilesN = Np >> 6;
    int tm = (wave / tilesN) << 6;
    int tn = (wave % tilesN) << 6;
    if (tm >= M) return;                 // wave-uniform: EXEC stays all-ones

    int rl = lane & 15;
    int kc = (lane >> 4) << 3;           // per-lane K chunk base (0 or 8)

    gcu16 Ag = (gcu16)A;
    gcu16 Bg = (gcu16)B;
    gcu16 pa[4];
    gcu16 pb[4];
#pragma unroll
    for (int i = 0; i < 4; ++i)
        pa[i] = Ag + (size_t)(tm + 16 * i + rl) * lda + kc;
#pragma unroll
    for (int j = 0; j < 4; ++j)
        pb[j] = Bg + (size_t)(tn + 16 * j + rl) * ldb + kc;

    f32x8 acc[4][4] = {};
    for (int k0 = 0; k0 < K; k0 += 32) {
        bf16x16 af[4], bfr[4];
#pragma unroll
        for (int i = 0; i < 4; ++i) { af[i]  = frag_at(pa[i]); pa[i] += 32; }
#pragma unroll
        for (int j = 0; j < 4; ++j) { bfr[j] = frag_at(pb[j]); pb[j] += 32; }
#pragma unroll
        for (int i = 0; i < 4; ++i)
#pragma unroll
            for (int j = 0; j < 4; ++j)
                acc[i][j] = wmma_bf16(af[i], bfr[j], acc[i][j]);
    }

    // C/D layout: lane gives N = lane&15; element e gives M = e + 8*(lane>>4)
    int nl = lane & 15;
    int mb = (lane >> 4) << 3;
#pragma unroll
    for (int i = 0; i < 4; ++i)
#pragma unroll
        for (int j = 0; j < 4; ++j) {
            int n = tn + 16 * j + nl;
            float bv = 0.f;
            if (bias && n < nbias) bv = bias[n];
#pragma unroll
            for (int e = 0; e < 8; ++e) {
                int m = tm + 16 * i + mb + e;
                float v = acc[i][j][e] + bv;
                if (do_relu) v = fmaxf(v, 0.f);
                C[(size_t)m * ldc + n] = f32_to_bf16(v);
            }
        }
}

// ---------- final GEMM: out = headA @ dep^T + rowt[i] + colt[j] + b_bil ----
__global__ __launch_bounds__(256)
void gemm_final(const u16* __restrict__ A, const u16* __restrict__ B,
                const float* __restrict__ rowt, const float* __restrict__ colt,
                const float* __restrict__ bbil, float* __restrict__ C,
                int M, int Np, int K, int lda, int ldb, int ldc) {
    int wave = (blockIdx.x * blockDim.x + threadIdx.x) >> 5;
    int lane = threadIdx.x & 31;
    int tilesN = Np >> 6;
    int tm = (wave / tilesN) << 6;
    int tn = (wave % tilesN) << 6;
    if (tm >= M) return;

    int rl = lane & 15;
    int kc = (lane >> 4) << 3;

    gcu16 Ag = (gcu16)A;
    gcu16 Bg = (gcu16)B;
    gcu16 pa[4];
    gcu16 pb[4];
#pragma unroll
    for (int i = 0; i < 4; ++i)
        pa[i] = Ag + (size_t)(tm + 16 * i + rl) * lda + kc;
#pragma unroll
    for (int j = 0; j < 4; ++j)
        pb[j] = Bg + (size_t)(tn + 16 * j + rl) * ldb + kc;

    f32x8 acc[4][4] = {};
    for (int k0 = 0; k0 < K; k0 += 32) {
        bf16x16 af[4], bfr[4];
#pragma unroll
        for (int i = 0; i < 4; ++i) { af[i]  = frag_at(pa[i]); pa[i] += 32; }
#pragma unroll
        for (int j = 0; j < 4; ++j) { bfr[j] = frag_at(pb[j]); pb[j] += 32; }
#pragma unroll
        for (int i = 0; i < 4; ++i)
#pragma unroll
            for (int j = 0; j < 4; ++j)
                acc[i][j] = wmma_bf16(af[i], bfr[j], acc[i][j]);
    }

    float bb = bbil[0];
    int nl = lane & 15;
    int mb = (lane >> 4) << 3;
#pragma unroll
    for (int i = 0; i < 4; ++i) {
        float rv[8];
#pragma unroll
        for (int e = 0; e < 8; ++e) rv[e] = rowt[tm + 16 * i + mb + e];
#pragma unroll
        for (int j = 0; j < 4; ++j) {
            int n = tn + 16 * j + nl;
            float cn = colt[n] + bb;
#pragma unroll
            for (int e = 0; e < 8; ++e) {
                int m = tm + 16 * i + mb + e;
                C[(size_t)m * ldc + n] = acc[i][j][e] + rv[e] + cn;
            }
        }
    }
}

// ---------- per-row dot with `extra` ----------
__global__ void rowdot(const u16* __restrict__ H, const float* __restrict__ extra,
                       float* __restrict__ out, int rows, int cols, int ldh) {
    int r = blockIdx.x * blockDim.x + threadIdx.x;
    if (r >= rows) return;
    float acc = 0.f;
    for (int j = 0; j < cols; ++j)
        acc += bf16_to_f32(H[(size_t)r * ldh + j]) * extra[j];
    out[r] = acc;
}

// ---------------------------------------------------------------------------
extern "C" void kernel_launch(void* const* d_in, const int* in_sizes, int n_in,
                              void* d_out, int out_size, void* d_ws, size_t ws_size,
                              hipStream_t stream) {
    const int N = 8192, IN = 768, H = 500, Hp = 512;

    const float* x    = (const float*)d_in[0];
    const float* W1h  = (const float*)d_in[1];
    const float* b1h  = (const float*)d_in[2];
    const float* W2h  = (const float*)d_in[3];
    const float* b2h  = (const float*)d_in[4];
    const float* W1d  = (const float*)d_in[5];
    const float* b1d  = (const float*)d_in[6];
    const float* W2d  = (const float*)d_in[7];
    const float* b2d  = (const float*)d_in[8];
    const float* Amat = (const float*)d_in[9];
    const float* bbil = (const float*)d_in[10];
    const float* extra= (const float*)d_in[11];

    char* ws = (char*)d_ws;
    size_t off = 0;
    auto take = [&](size_t bytes) -> void* {
        void* p = (void*)(ws + off);
        off = (off + bytes + 255) & ~(size_t)255;
        return p;
    };

    u16* xb    = (u16*)take((size_t)N * IN * 2);
    u16* w1hb  = (u16*)take((size_t)Hp * IN * 2);
    u16* w1db  = (u16*)take((size_t)Hp * IN * 2);
    u16* w2hb  = (u16*)take((size_t)Hp * Hp * 2);
    u16* w2db  = (u16*)take((size_t)Hp * Hp * 2);
    u16* atb   = (u16*)take((size_t)Hp * Hp * 2);   // A^T padded
    u16* h1h   = (u16*)take((size_t)N * Hp * 2);    // reused later for headA
    u16* h1d   = (u16*)take((size_t)N * Hp * 2);
    u16* headb = (u16*)take((size_t)N * Hp * 2);
    u16* depb  = (u16*)take((size_t)N * Hp * 2);
    float* rowt = (float*)take((size_t)N * 4);
    float* colt = (float*)take((size_t)N * 4);
    (void)ws_size; (void)in_sizes; (void)n_in; (void)out_size;

    // ---- convert / pad inputs to bf16 ----
    {
        int t = N * IN;
        cvt_pad<<<(t + 255) / 256, 256, 0, stream>>>(x, xb, N, IN, N, IN, 0);
        t = Hp * IN;
        cvt_pad<<<(t + 255) / 256, 256, 0, stream>>>(W1h, w1hb, H, IN, Hp, IN, 0);
        cvt_pad<<<(t + 255) / 256, 256, 0, stream>>>(W1d, w1db, H, IN, Hp, IN, 0);
        t = Hp * Hp;
        cvt_pad<<<(t + 255) / 256, 256, 0, stream>>>(W2h, w2hb, H, H, Hp, Hp, 0);
        cvt_pad<<<(t + 255) / 256, 256, 0, stream>>>(W2d, w2db, H, H, Hp, Hp, 0);
        cvt_pad<<<(t + 255) / 256, 256, 0, stream>>>(Amat, atb, H, H, Hp, Hp, 1);
    }

    // ---- MLPs + bilinear projection ----
    {
        int waves = (N / 64) * (Hp / 64);               // 1024
        int blocks = waves * 32 / 256;                  // 128
        gemm_bf16<<<blocks, 256, 0, stream>>>(xb, w1hb, b1h, H, 1, h1h,
                                              N, Hp, IN, IN, IN, Hp);
        gemm_bf16<<<blocks, 256, 0, stream>>>(xb, w1db, b1d, H, 1, h1d,
                                              N, Hp, IN, IN, IN, Hp);
        gemm_bf16<<<blocks, 256, 0, stream>>>(h1h, w2hb, b2h, H, 0, headb,
                                              N, Hp, Hp, Hp, Hp, Hp);
        gemm_bf16<<<blocks, 256, 0, stream>>>(h1d, w2db, b2d, H, 0, depb,
                                              N, Hp, Hp, Hp, Hp, Hp);
        // headA = head @ A  (B = A^T, row-contiguous K); reuse h1h buffer
        gemm_bf16<<<blocks, 256, 0, stream>>>(headb, atb, nullptr, 0, 0, h1h,
                                              N, Hp, Hp, Hp, Hp, Hp);
    }

    // ---- rank-1 terms ----
    rowdot<<<N / 256, 256, 0, stream>>>(headb, extra, rowt, N, H, Hp);
    rowdot<<<N / 256, 256, 0, stream>>>(depb,  extra, colt, N, H, Hp);

    // ---- final: out = headA @ dep^T + rowt + colt + b_bil ----
    {
        int waves = (N / 64) * (N / 64);                // 16384
        int blocks = waves * 32 / 256;                  // 2048
        gemm_final<<<blocks, 256, 0, stream>>>(h1h, depb, rowt, colt, bbil,
                                               (float*)d_out,
                                               N, N, Hp, Hp, Hp, N);
    }
}
